// SparseMoELanguageModel_69526930588344
// MI455X (gfx1250) — matrix operand order repaired and checked by reading, not compiled, not run on hardware
//
#include <hip/hip_runtime.h>
#include <hip/hip_bf16.h>
#include <math.h>

typedef __attribute__((ext_vector_type(16))) _Float16 v16h;
typedef __attribute__((ext_vector_type(8)))  _Float16 v8h;
typedef __attribute__((ext_vector_type(8)))  float    v8f;

// ---- model constants ----
static constexpr int Bc = 1, Tc = 2048, Dc = 1024;
static constexpr int Hc = 16, HSc = 64, HALFc = 32;
static constexpr int Ec = 8, Kc = 2, Lc = 2;
static constexpr int CAPc = Bc * Tc * Kc / Ec;   // 512
static constexpr int Nt = Bc * Tc;                // tokens
static constexpr int D4 = 4 * Dc;                 // 4096

// ---- GEMM tiling ----
static constexpr int TM = 4, TN = 2;              // 16x16 tiles per wave -> 64x32
static constexpr int WAVES_M = 2, WAVES_N = 2;    // 4 waves/block, 2x2
static constexpr int BM = 16 * TM * WAVES_M;      // 128 rows per block
static constexpr int BN = 16 * TN * WAVES_N;      // 64 cols per block
static constexpr int GEMM_THREADS = 32 * WAVES_M * WAVES_N;   // 128

// =====================================================================
// Register-blocked WMMA GEMM:
//   C[M,N] = scale * (A[M,Kd] @ W[N,Kd]^T) (+bias) (relu)
// A, W f16 row-major; C f32 with leading dim ldc. Kd multiple of 64.
// grid: (N/BN, M/BM, batch); block: GEMM_THREADS.
// Each wave computes a TM x TN grid of 16x16 WMMA tiles (64x32): per
// 32-deep k-step it loads TM+TN fragments and issues TM*TN v_wmma ops.
// k-loop unrolled 2x so the scheduler can overlap loads with WMMAs
// without forcing both iterations' fragments live (no spills).
// Fragment layouts: A and B both contiguous-K (group g holds k+16g..k+16g+15).
// =====================================================================
template<bool BIAS, bool RELU, bool CAUSAL>
__global__ void gemm_wmma_kernel(const _Float16* __restrict__ A,
                                 const _Float16* __restrict__ Wt,
                                 const float* __restrict__ bias,
                                 float* __restrict__ C,
                                 int M, int N, int Kd, int ldc, float scale,
                                 long long sA, long long sW, long long sB, long long sC)
{
    A  += (long long)blockIdx.z * sA;
    Wt += (long long)blockIdx.z * sW;
    if (BIAS) bias += (long long)blockIdx.z * sB;
    C  += (long long)blockIdx.z * sC;

    const int wave = threadIdx.x >> 5;
    const int lane = threadIdx.x & 31;
    const int wm = wave >> 1, wn = wave & 1;
    const int wrow0 = blockIdx.y * BM + wm * (16 * TM);
    const int wcol0 = blockIdx.x * BN + wn * (16 * TN);

    // wave-uniform skips keep EXEC all-ones for WMMA
    if (CAUSAL && wcol0 >= wrow0 + 16 * TM) return;  // wave tile fully above diagonal
    if (wrow0 >= M || wcol0 >= N) return;

    const int g = lane >> 4;      // lane group 0/1
    const int r = lane & 15;

    const _Float16* arow[TM];
    const _Float16* brow[TN];
#pragma unroll
    for (int m = 0; m < TM; ++m)
        arow[m] = A + (size_t)(wrow0 + m * 16 + r) * Kd;
#pragma unroll
    for (int n = 0; n < TN; ++n)
        brow[n] = Wt + (size_t)(wcol0 + n * 16 + r) * Kd;

    v8f acc[TM][TN] = {};

#pragma unroll 2
    for (int k = 0; k < Kd; k += 32) {
        v16h af[TM], bf[TN];
#pragma unroll
        for (int m = 0; m < TM; ++m) {
            // A fragment (16x32 f16): group g holds K = k+16g .. k+16g+15
            af[m] = *(const v16h*)(arow[m] + k + g * 16);
        }
#pragma unroll
        for (int n = 0; n < TN; ++n) {
            // B fragment (32x16 f16): group g holds K = k+16g .. k+16g+15
            bf[n] = *(const v16h*)(brow[n] + k + g * 16);
        }
#pragma unroll
        for (int m = 0; m < TM; ++m)
#pragma unroll
            for (int n = 0; n < TN; ++n)
                acc[m][n] = __builtin_amdgcn_wmma_f32_16x16x32_f16(
                                false, af[m], false, bf[n], (short)0,
                                acc[m][n], false, false);
    }

#pragma unroll
    for (int m = 0; m < TM; ++m) {
#pragma unroll
        for (int n = 0; n < TN; ++n) {
            const int col = wcol0 + n * 16 + r;
            const float bv = BIAS ? bias[col] : 0.0f;
            float* crow = C + (size_t)(wrow0 + m * 16 + g * 8) * ldc + col;
#pragma unroll
            for (int i = 0; i < 8; ++i) {
                float v = acc[m][n][i] * scale + bv;
                if (RELU) v = fmaxf(v, 0.0f);
                crow[(size_t)i * ldc] = v;
            }
        }
    }
}

// =====================================================================
// Elementwise / helper kernels
// =====================================================================
__global__ void embed_kernel(const int* __restrict__ ids,
                             const float* __restrict__ tok,
                             const float* __restrict__ pos,
                             float* __restrict__ x)
{
    int idx = blockIdx.x * blockDim.x + threadIdx.x;
    if (idx >= Nt * Dc) return;
    int d = idx % Dc, t = idx / Dc;
    x[idx] = tok[(size_t)ids[t] * Dc + d] + pos[idx];
}

__global__ void cvt_f32_f16_kernel(const float* __restrict__ src,
                                   _Float16* __restrict__ dst, long long n)
{
    long long i = (long long)blockIdx.x * blockDim.x + threadIdx.x;
    if (i < n) dst[i] = (_Float16)src[i];
}

__global__ void add_kernel(float* __restrict__ x, const float* __restrict__ y, int n)
{
    int i = blockIdx.x * blockDim.x + threadIdx.x;
    if (i < n) x[i] += y[i];
}

__global__ void fill_i32_kernel(int* __restrict__ p, int v, int n)
{
    int i = blockIdx.x * blockDim.x + threadIdx.x;
    if (i < n) p[i] = v;
}

__global__ void layernorm_kernel(const float* __restrict__ x,
                                 const float* __restrict__ w,
                                 const float* __restrict__ bb,
                                 float* __restrict__ o32,
                                 _Float16* __restrict__ o16)
{
    const int row = blockIdx.x;
    const float* xr = x + (size_t)row * Dc;
    __shared__ float s1[256], s2[256];
    float a = 0.f, q = 0.f;
    for (int i = threadIdx.x; i < Dc; i += 256) { float v = xr[i]; a += v; q += v * v; }
    s1[threadIdx.x] = a; s2[threadIdx.x] = q;
    __syncthreads();
    for (int st = 128; st > 0; st >>= 1) {
        if (threadIdx.x < st) {
            s1[threadIdx.x] += s1[threadIdx.x + st];
            s2[threadIdx.x] += s2[threadIdx.x + st];
        }
        __syncthreads();
    }
    const float mean = s1[0] / Dc;
    const float var  = s2[0] / Dc - mean * mean;
    const float inv  = rsqrtf(var + 1e-5f);
    for (int i = threadIdx.x; i < Dc; i += 256) {
        float v = (xr[i] - mean) * inv * w[i] + bb[i];
        if (o32) o32[(size_t)row * Dc + i] = v;
        if (o16) o16[(size_t)row * Dc + i] = (_Float16)v;
    }
}

// qkv[T][3][H][HS] f32 -> rope'd q,k head-major f16 + v transposed per head
__global__ void rope_split_kernel(const float* __restrict__ qkv,
                                  _Float16* __restrict__ q,
                                  _Float16* __restrict__ k,
                                  _Float16* __restrict__ vt)
{
    int idx = blockIdx.x * blockDim.x + threadIdx.x;    // (h, t, i<HALF)
    if (idx >= Hc * Tc * HALFc) return;
    int i = idx % HALFc;
    int t = (idx / HALFc) % Tc;
    int h = idx / (HALFc * Tc);
    float inv = __expf(-logf(10000.0f) * (2.0f * i) / (float)HSc);
    float s = __sinf((float)t * inv), c = __cosf((float)t * inv);
    const float* base = qkv + (size_t)t * 3 * Dc;
    float q1 = base[h * HSc + i],          q2 = base[h * HSc + i + HALFc];
    float k1 = base[Dc + h * HSc + i],     k2 = base[Dc + h * HSc + i + HALFc];
    float v1 = base[2 * Dc + h * HSc + i], v2 = base[2 * Dc + h * HSc + i + HALFc];
    size_t o = ((size_t)h * Tc + t) * HSc;
    q[o + i]         = (_Float16)(q1 * c - q2 * s);
    q[o + i + HALFc] = (_Float16)(q2 * c + q1 * s);
    k[o + i]         = (_Float16)(k1 * c - k2 * s);
    k[o + i + HALFc] = (_Float16)(k2 * c + k1 * s);
    vt[((size_t)h * HSc + i) * Tc + t]         = (_Float16)v1;
    vt[((size_t)h * HSc + i + HALFc) * Tc + t] = (_Float16)v2;
}

__global__ void softmax_causal_kernel(const float* __restrict__ S,
                                      _Float16* __restrict__ P)
{
    const int row = blockIdx.x;
    const int len = row + 1;
    const float* sr = S + (size_t)row * Tc;
    _Float16* pr = P + (size_t)row * Tc;
    __shared__ float red[256];
    float m = -1e30f;
    for (int j = threadIdx.x; j < len; j += 256) m = fmaxf(m, sr[j]);
    red[threadIdx.x] = m; __syncthreads();
    for (int st = 128; st > 0; st >>= 1) {
        if (threadIdx.x < st) red[threadIdx.x] = fmaxf(red[threadIdx.x], red[threadIdx.x + st]);
        __syncthreads();
    }
    const float mx = red[0]; __syncthreads();
    float sum = 0.f;
    for (int j = threadIdx.x; j < len; j += 256) sum += __expf(sr[j] - mx);
    red[threadIdx.x] = sum; __syncthreads();
    for (int st = 128; st > 0; st >>= 1) {
        if (threadIdx.x < st) red[threadIdx.x] += red[threadIdx.x + st];
        __syncthreads();
    }
    const float inv = 1.0f / red[0];
    for (int j = threadIdx.x; j < Tc; j += 256)
        pr[j] = (j < len) ? (_Float16)(__expf(sr[j] - mx) * inv) : (_Float16)0.0f;
}

__global__ void router_kernel(const float* __restrict__ xln,
                              const float* __restrict__ rw, const float* __restrict__ rb,
                              const float* __restrict__ nw, const float* __restrict__ nb,
                              const float* __restrict__ noise,
                              float* __restrict__ noisy)
{
    int idx = blockIdx.x * blockDim.x + threadIdx.x;   // t*E + e
    if (idx >= Nt * Ec) return;
    int e = idx % Ec, t = idx / Ec;
    const float* xr = xln + (size_t)t * Dc;
    const float* r  = rw + (size_t)e * Dc;
    const float* n  = nw + (size_t)e * Dc;
    float lr = 0.f, ln_ = 0.f;
    for (int i = 0; i < Dc; ++i) { lr += xr[i] * r[i]; ln_ += xr[i] * n[i]; }
    lr += rb[e]; ln_ += nb[e];
    float sp = (ln_ > 20.f) ? ln_ : log1pf(__expf(ln_));   // softplus
    noisy[idx] = lr + noise[idx] * sp;
}

__global__ void topk_gate_kernel(const float* __restrict__ noisy,
                                 int* __restrict__ topi, float* __restrict__ gate)
{
    int t = blockIdx.x * blockDim.x + threadIdx.x;
    if (t >= Nt) return;
    const float* nr = noisy + (size_t)t * Ec;
    int i0 = 0; float v0 = nr[0];
    for (int e = 1; e < Ec; ++e) if (nr[e] > v0) { v0 = nr[e]; i0 = e; }
    int i1 = (i0 == 0) ? 1 : 0; float v1 = nr[i1];
    for (int e = 0; e < Ec; ++e) { if (e == i0 || e == i1) continue; if (nr[e] > v1) { v1 = nr[e]; i1 = e; } }
    float m = fmaxf(v0, v1);
    float e0 = __expf(v0 - m), e1 = __expf(v1 - m);
    float inv = 1.0f / (e0 + e1);
    topi[t * Kc] = i0; topi[t * Kc + 1] = i1;
    gate[t * Kc] = e0 * inv; gate[t * Kc + 1] = e1 * inv;
}

// serial capacity assignment, matches reference cumsum ordering over (token, k)
__global__ void assign_kernel(const int* __restrict__ topi,
                              int* __restrict__ pos, int* __restrict__ slot_src)
{
    if (blockIdx.x != 0 || threadIdx.x != 0) return;
    int cnt[Ec];
    for (int e = 0; e < Ec; ++e) cnt[e] = 0;
    for (int i = 0; i < Nt * Kc; ++i) {
        int e = topi[i];
        int r = cnt[e]++;
        if (r < CAPc) { pos[i] = e * CAPc + r; slot_src[e * CAPc + r] = i / Kc; }
        else pos[i] = -1;
    }
}

__global__ void dispatch_kernel(const float* __restrict__ xln,
                                const int* __restrict__ slot_src,
                                _Float16* __restrict__ de)
{
    long long idx = (long long)blockIdx.x * blockDim.x + threadIdx.x;  // slot*D + d
    if (idx >= (long long)Ec * CAPc * Dc) return;
    int d = (int)(idx % Dc);
    int slot = (int)(idx / Dc);
    int tok = slot_src[slot];
    de[idx] = (tok >= 0) ? (_Float16)xln[(size_t)tok * Dc + d] : (_Float16)0.0f;
}

__global__ void combine_kernel(const float* __restrict__ eo,
                               const int* __restrict__ pos,
                               const float* __restrict__ gate,
                               float* __restrict__ x)
{
    int idx = blockIdx.x * blockDim.x + threadIdx.x;   // t*D + d
    if (idx >= Nt * Dc) return;
    int d = idx % Dc, t = idx / Dc;
    float acc = 0.f;
#pragma unroll
    for (int kk = 0; kk < Kc; ++kk) {
        int p = pos[t * Kc + kk];
        if (p >= 0) acc += gate[t * Kc + kk] * eo[(size_t)p * Dc + d];
    }
    x[idx] += acc;
}

// =====================================================================
// Host orchestration
// =====================================================================
static inline int cdiv(long long a, int b) { return (int)((a + b - 1) / b); }

extern "C" void kernel_launch(void* const* d_in, const int* in_sizes, int n_in,
                              void* d_out, int out_size, void* d_ws, size_t ws_size,
                              hipStream_t stream)
{
    const int*   ids      = (const int*)  d_in[0];
    const float* noise    = (const float*)d_in[1];
    const float* tok_emb  = (const float*)d_in[2];
    const float* pos_emb  = (const float*)d_in[3];
    const float* ln1_w    = (const float*)d_in[4];
    const float* ln1_b    = (const float*)d_in[5];
    const float* ln2_w    = (const float*)d_in[6];
    const float* ln2_b    = (const float*)d_in[7];
    const float* qkv_w    = (const float*)d_in[8];
    const float* out_w    = (const float*)d_in[9];
    const float* router_w = (const float*)d_in[10];
    const float* router_b = (const float*)d_in[11];
    const float* noise_w  = (const float*)d_in[12];
    const float* noise_b  = (const float*)d_in[13];
    const float* e_w1     = (const float*)d_in[14];
    const float* e_b1     = (const float*)d_in[15];
    const float* e_w2     = (const float*)d_in[16];
    const float* e_b2     = (const float*)d_in[17];
    const float* lnf_w    = (const float*)d_in[18];
    const float* lnf_b    = (const float*)d_in[19];

    char* ws = (char*)d_ws;
    size_t off = 0;
    auto carve = [&](size_t bytes) -> char* {
        char* p = ws + off;
        off += (bytes + 255) & ~(size_t)255;
        return p;
    };

    float*    x        = (float*)   carve((size_t)Nt * Dc * 4);
    float*    xln      = (float*)   carve((size_t)Nt * Dc * 4);
    _Float16* xln_h    = (_Float16*)carve((size_t)Nt * Dc * 2);
    float*    qkv      = (float*)   carve((size_t)Nt * 3 * Dc * 4);
    _Float16* q_h      = (_Float16*)carve((size_t)Hc * Tc * HSc * 2);
    _Float16* k_h      = (_Float16*)carve((size_t)Hc * Tc * HSc * 2);
    _Float16* vt_h     = (_Float16*)carve((size_t)Hc * HSc * Tc * 2);
    float*    scores   = (float*)   carve((size_t)Tc * Tc * 4);
    _Float16* p_h      = (_Float16*)carve((size_t)Tc * Tc * 2);
    float*    attnout  = (float*)   carve((size_t)Nt * Dc * 4);
    _Float16* attn_h   = (_Float16*)carve((size_t)Nt * Dc * 2);
    float*    proj     = (float*)   carve((size_t)Nt * Dc * 4);
    _Float16* wq_h     = (_Float16*)carve((size_t)3 * Dc * Dc * 2);
    _Float16* wo_h     = (_Float16*)carve((size_t)Dc * Dc * 2);
    _Float16* w1_h     = (_Float16*)carve((size_t)Ec * D4 * Dc * 2);
    _Float16* w2_h     = (_Float16*)carve((size_t)Ec * Dc * D4 * 2);
    float*    noisy    = (float*)   carve((size_t)Nt * Ec * 4);
    int*      topi     = (int*)     carve((size_t)Nt * Kc * 4);
    float*    gate     = (float*)   carve((size_t)Nt * Kc * 4);
    int*      posb     = (int*)     carve((size_t)Nt * Kc * 4);
    int*      slot_src = (int*)     carve((size_t)Ec * CAPc * 4);
    _Float16* de_h     = (_Float16*)carve((size_t)Ec * CAPc * Dc * 2);
    float*    hbuf     = (float*)   carve((size_t)Ec * CAPc * D4 * 4);
    _Float16* h_h      = (_Float16*)carve((size_t)Ec * CAPc * D4 * 2);
    float*    eo       = (float*)   carve((size_t)Ec * CAPc * Dc * 4);

    const int BT = 256;
    auto cvt = [&](const float* s, _Float16* d, long long n) {
        cvt_f32_f16_kernel<<<cdiv(n, BT), BT, 0, stream>>>(s, d, n);
    };

    // x = tok_emb[ids] + pos_emb
    embed_kernel<<<cdiv((long long)Nt * Dc, BT), BT, 0, stream>>>(ids, tok_emb, pos_emb, x);

    for (int l = 0; l < Lc; ++l) {
        // ---- layer weights -> f16 ----
        cvt(qkv_w + (size_t)l * 3 * Dc * Dc, wq_h, (long long)3 * Dc * Dc);
        cvt(out_w + (size_t)l * Dc * Dc,     wo_h, (long long)Dc * Dc);
        cvt(e_w1 + (size_t)l * Ec * D4 * Dc, w1_h, (long long)Ec * D4 * Dc);
        cvt(e_w2 + (size_t)l * Ec * Dc * D4, w2_h, (long long)Ec * Dc * D4);

        // ---- attention ----
        layernorm_kernel<<<Nt, 256, 0, stream>>>(x, ln1_w + l * Dc, ln1_b + l * Dc,
                                                 (float*)nullptr, xln_h);
        // qkv = ln1(x) @ qkv_w^T   [T, 3D]
        gemm_wmma_kernel<false,false,false>
            <<<dim3(3 * Dc / BN, Tc / BM, 1), GEMM_THREADS, 0, stream>>>(
                xln_h, wq_h, nullptr, qkv, Tc, 3 * Dc, Dc, 3 * Dc, 1.0f, 0, 0, 0, 0);

        rope_split_kernel<<<cdiv((long long)Hc * Tc * HALFc, BT), BT, 0, stream>>>(
            qkv, q_h, k_h, vt_h);

        for (int h = 0; h < Hc; ++h) {
            // S = (Q_h @ K_h^T) / 8 with causal wave-tile skip
            gemm_wmma_kernel<false,false,true>
                <<<dim3(Tc / BN, Tc / BM, 1), GEMM_THREADS, 0, stream>>>(
                    q_h + (size_t)h * Tc * HSc, k_h + (size_t)h * Tc * HSc, nullptr,
                    scores, Tc, Tc, HSc, Tc, 0.125f, 0, 0, 0, 0);
            softmax_causal_kernel<<<Tc, 256, 0, stream>>>(scores, p_h);
            // AV = P @ Vt^T  -> attnout[:, h*64 : h*64+64]
            gemm_wmma_kernel<false,false,false>
                <<<dim3(HSc / BN, Tc / BM, 1), GEMM_THREADS, 0, stream>>>(
                    p_h, vt_h + (size_t)h * HSc * Tc, nullptr,
                    attnout + h * HSc, Tc, HSc, Tc, Dc, 1.0f, 0, 0, 0, 0);
        }

        cvt(attnout, attn_h, (long long)Nt * Dc);
        gemm_wmma_kernel<false,false,false>
            <<<dim3(Dc / BN, Tc / BM, 1), GEMM_THREADS, 0, stream>>>(
                attn_h, wo_h, nullptr, proj, Tc, Dc, Dc, Dc, 1.0f, 0, 0, 0, 0);
        add_kernel<<<cdiv((long long)Nt * Dc, BT), BT, 0, stream>>>(x, proj, Nt * Dc);

        // ---- MoE ----
        layernorm_kernel<<<Nt, 256, 0, stream>>>(x, ln2_w + l * Dc, ln2_b + l * Dc,
                                                 xln, (_Float16*)nullptr);
        router_kernel<<<cdiv((long long)Nt * Ec, BT), BT, 0, stream>>>(
            xln, router_w + (size_t)l * Ec * Dc, router_b + l * Ec,
            noise_w + (size_t)l * Ec * Dc, noise_b + l * Ec,
            noise + (size_t)l * Nt * Ec, noisy);
        topk_gate_kernel<<<cdiv(Nt, BT), BT, 0, stream>>>(noisy, topi, gate);
        fill_i32_kernel<<<cdiv(Ec * CAPc, BT), BT, 0, stream>>>(slot_src, -1, Ec * CAPc);
        assign_kernel<<<1, 32, 0, stream>>>(topi, posb, slot_src);
        dispatch_kernel<<<cdiv((long long)Ec * CAPc * Dc, BT), BT, 0, stream>>>(
            xln, slot_src, de_h);

        // h = relu(de @ w1^T + b1)  batched over experts
        gemm_wmma_kernel<true,true,false>
            <<<dim3(D4 / BN, CAPc / BM, Ec), GEMM_THREADS, 0, stream>>>(
                de_h, w1_h, e_b1 + (size_t)l * Ec * D4, hbuf,
                CAPc, D4, Dc, D4, 1.0f,
                (long long)CAPc * Dc, (long long)D4 * Dc, (long long)D4,
                (long long)CAPc * D4);
        cvt(hbuf, h_h, (long long)Ec * CAPc * D4);
        // eo = h @ w2^T + b2
        gemm_wmma_kernel<true,false,false>
            <<<dim3(Dc / BN, CAPc / BM, Ec), GEMM_THREADS, 0, stream>>>(
                h_h, w2_h, e_b2 + (size_t)l * Ec * Dc, eo,
                CAPc, Dc, D4, Dc, 1.0f,
                (long long)CAPc * D4, (long long)Dc * D4, (long long)Dc,
                (long long)CAPc * Dc);
        combine_kernel<<<cdiv((long long)Nt * Dc, BT), BT, 0, stream>>>(eo, posb, gate, x);
    }

    // final layernorm -> d_out (f32)
    layernorm_kernel<<<Nt, 256, 0, stream>>>(x, lnf_w, lnf_b, (float*)d_out,
                                             (_Float16*)nullptr);
    (void)in_sizes; (void)n_in; (void)out_size; (void)ws_size;
}
